// LongformerBartEncoderLayer_73315091743432
// MI455X (gfx1250) — matrix operand order repaired and checked by reading, not compiled, not run on hardware
//
#include <hip/hip_runtime.h>
#include <math.h>

// ---------------------------------------------------------------------------
// Longformer/BART encoder layer for MI455X (gfx1250, wave32, WMMA + TDM).
// All matmuls use v_wmma_f32_16x16x32_f16 (fp16 in, fp32 accumulate).
// GEMM A and B tiles are both staged into LDS by the Tensor Data Mover,
// double-buffered and software-pipelined against the WMMAs (TENSORcnt).
// Weights are pre-transposed to [N,K] so the B tile is a plain 2D copy.
// ---------------------------------------------------------------------------

typedef __attribute__((ext_vector_type(16))) _Float16 v16h;
typedef __attribute__((ext_vector_type(8)))  float    v8f;
typedef __attribute__((ext_vector_type(4)))  unsigned v4u;
typedef __attribute__((ext_vector_type(8)))  int      v8i;
typedef __attribute__((ext_vector_type(4)))  int      v4i;

#define B_   2
#define S_   4096
#define D_   1024
#define H_   16
#define HD_  64
#define FFN_ 4096
#define W_   256            // one-sided window
#define NB_  (S_ / W_)      // 16 query blocks
#define NEGF (-1e9f)

#if defined(__gfx1250__) && __has_builtin(__builtin_amdgcn_tensor_load_to_lds) && \
    __has_builtin(__builtin_amdgcn_s_wait_tensorcnt)
#define HAVE_TDM 1
#else
#define HAVE_TDM 0
#endif

// 16-bit A-operand k order used by this code (contiguous k, B loads use the same):
// lane half 'hi' (lane>>4), packed half index h (0..15) -> K index in [0,32)
__device__ __forceinline__ int kmapA(int h, int hi) {
  return 16 * hi + h;
}

#if HAVE_TDM
// Issue a TDM 2D tile load (f16 elements): rows x row_halves tile from a
// row-major tensor with row stride tensor_row_halves, into LDS at lds_byte_off.
// Descriptor layout per CDNA5 ISA.
__device__ __forceinline__ void tdm_load_2d_f16(unsigned lds_byte_off,
                                                const void* gaddr,
                                                int rows, int row_halves,
                                                int tensor_row_halves,
                                                int tensor_rows) {
  unsigned long long ga = (unsigned long long)gaddr;
  v4u g0;
  g0[0] = 1u;                                     // count=1 valid descriptor
  g0[1] = lds_byte_off;                           // lds_addr (bytes)
  g0[2] = (unsigned)(ga & 0xFFFFFFFFu);           // global_addr[31:0]
  g0[3] = (unsigned)((ga >> 32) & 0x01FFFFFFu) | (2u << 30);  // [56:32] | type=2
  v8i g1;
  g1[0] = (int)(1u << 16);                        // workgroup_mask=0, data_size=1 (2B)
  g1[1] = (int)(((unsigned)tensor_row_halves & 0xFFFFu) << 16);      // tensor_dim0 lo
  g1[2] = (int)((((unsigned)tensor_row_halves >> 16) & 0xFFFFu) |
                (((unsigned)tensor_rows & 0xFFFFu) << 16));          // dim0 hi | dim1 lo
  g1[3] = (int)(((((unsigned)tensor_rows) >> 16) & 0xFFFFu) |
                (((unsigned)row_halves & 0xFFFFu) << 16));           // dim1 hi | tile_dim0
  g1[4] = (int)((unsigned)rows & 0xFFFFu);        // tile_dim1 | tile_dim2=0
  g1[5] = (int)tensor_row_halves;                 // tensor_dim0_stride[31:0]
  g1[6] = 0;                                      // stride hi | dim1_stride lo
  g1[7] = 0;                                      // dim1_stride hi
  v4i z4 = {0, 0, 0, 0};
#if __clang_major__ >= 23
  v8i z8 = {0, 0, 0, 0, 0, 0, 0, 0};
  __builtin_amdgcn_tensor_load_to_lds(g0, g1, z4, z4, z8, 0);
#else
  __builtin_amdgcn_tensor_load_to_lds(g0, g1, z4, z4, 0);
#endif
}
#endif  // HAVE_TDM

// ---------------------------------------------------------------------------
// fp32 -> fp16 conversion (layout-preserving)
// ---------------------------------------------------------------------------
__global__ __launch_bounds__(256) void cvt_f16_kernel(const float* __restrict__ in,
                                                      _Float16* __restrict__ out, int n) {
  int base = blockIdx.x * 1024 + threadIdx.x;
#pragma unroll
  for (int j = 0; j < 4; ++j) {
    int idx = base + j * 256;
    if (idx < n) out[idx] = (_Float16)in[idx];
  }
}

// ---------------------------------------------------------------------------
// fp32 [K,N] -> fp16 [N,K] transpose-convert (LDS 32x32 tile, both sides
// coalesced). K, N multiples of 32. Done once per weight so that GEMM B
// tiles become plain row-major 2D copies the TDM can move.
// ---------------------------------------------------------------------------
__global__ __launch_bounds__(256) void cvt_t_f16_kernel(const float* __restrict__ in,
                                                        _Float16* __restrict__ out,
                                                        int K, int N) {
  __shared__ float tile[32][33];
  const int tx = threadIdx.x & 31;
  const int ty = threadIdx.x >> 5;   // 0..7
  const int k0 = blockIdx.x * 32;
  const int n0 = blockIdx.y * 32;
#pragma unroll
  for (int i = 0; i < 32; i += 8)
    tile[ty + i][tx] = in[(size_t)(k0 + ty + i) * N + n0 + tx];
  __syncthreads();
#pragma unroll
  for (int i = 0; i < 32; i += 8)
    out[(size_t)(n0 + ty + i) * K + k0 + tx] = (_Float16)tile[tx][ty + i];
}

// ---------------------------------------------------------------------------
// Per-wave WMMA compute on one 32-deep K-slice from LDS.
// AsB: [128][32] row-major (rows = M), BsB: [128][32] row-major (rows = N).
// ---------------------------------------------------------------------------
__device__ __forceinline__ void wmma_tile_step(const _Float16* AsB, const _Float16* BsB,
                                               int wave_m, int wave_n, int nlo, int hi,
                                               v8f (&acc)[4][2]) {
  v16h a[4], b[2];
#pragma unroll
  for (int mt = 0; mt < 4; ++mt) {
    int row = wave_m + mt * 16 + nlo;
#pragma unroll
    for (int h = 0; h < 16; ++h) a[mt][h] = AsB[row * 32 + kmapA(h, hi)];
  }
#pragma unroll
  for (int nt = 0; nt < 2; ++nt) {
    int col = wave_n + nt * 16 + nlo;
#pragma unroll
    for (int h = 0; h < 16; ++h) b[nt][h] = BsB[col * 32 + 16 * hi + h];
  }
#pragma unroll
  for (int mt = 0; mt < 4; ++mt)
#pragma unroll
    for (int nt = 0; nt < 2; ++nt)
      acc[mt][nt] = __builtin_amdgcn_wmma_f32_16x16x32_f16(
          false, a[mt], false, b[nt], (short)0, acc[mt][nt], false, false);
}

// ---------------------------------------------------------------------------
// Tiled WMMA GEMM: C[M,N] = A[M,K](f16) * BT[N,K](f16)^T + bias, epilogue.
// 256 threads = 8 waves; tile 128x128; wave = 64x32 (4x2 WMMA tiles); K-step 32.
// TDM path: both LDS tiles are DMA'd by wave 0, double-buffered so the DMA of
// tile i+1 overlaps the WMMAs of tile i (s_wait_tensorcnt <= 2 pipelining).
// epi: 0 = f16 out, 1 = exact GELU -> f16 out, 2 = +resid(f32) -> f32 out
// M,N multiples of 128; K multiple of 32 (true for all calls here).
// ---------------------------------------------------------------------------
__global__ __launch_bounds__(256) void gemm_f16_kernel(
    const _Float16* __restrict__ A, const _Float16* __restrict__ BT,
    const float* __restrict__ bias, const float* __restrict__ resid,
    _Float16* __restrict__ out16, float* __restrict__ out32,
    int M, int N, int K, int epi) {
  __shared__ _Float16 As[2][128 * 32];   // [buf][row][k]
  __shared__ _Float16 Bs[2][128 * 32];   // [buf][n][k]

  const int tid  = threadIdx.x;
  const int lane = tid & 31;
  const int wv   = tid >> 5;
  const int nlo  = lane & 15;
  const int hi   = lane >> 4;
  const int gm0  = blockIdx.x * 128;
  const int gn0  = blockIdx.y * 128;
  const int wave_m = (wv & 1) * 64;
  const int wave_n = (wv >> 1) * 32;

  const v8f zf = {0.f, 0.f, 0.f, 0.f, 0.f, 0.f, 0.f, 0.f};
  v8f acc[4][2];
#pragma unroll
  for (int mt = 0; mt < 4; ++mt)
#pragma unroll
    for (int nt = 0; nt < 2; ++nt) acc[mt][nt] = zf;

#if HAVE_TDM
  const unsigned as_off = (unsigned)(uintptr_t)(void*)&As[0][0];
  const unsigned bs_off = (unsigned)(uintptr_t)(void*)&Bs[0][0];
  const int nk = K / 32;
  // prologue: DMA tiles for iteration 0 into buffer 0
  if (wv == 0) {
    tdm_load_2d_f16(as_off, A + (size_t)gm0 * K, 128, 32, K, M);
    tdm_load_2d_f16(bs_off, BT + (size_t)gn0 * K, 128, 32, K, N);
  }
  for (int it = 0; it < nk; ++it) {
    const int k0  = it * 32;
    const int buf = it & 1;
    if (wv == 0) {
      if (it + 1 < nk) {
        // DMA next tiles into the other buffer; previous-iteration compute on
        // that buffer finished before this iteration's issuing barrier.
        tdm_load_2d_f16(as_off + (buf ^ 1) * 8192, A + (size_t)gm0 * K + k0 + 32, 128, 32, K, M);
        tdm_load_2d_f16(bs_off + (buf ^ 1) * 8192, BT + (size_t)gn0 * K + k0 + 32, 128, 32, K, N);
        __builtin_amdgcn_s_wait_tensorcnt(2);   // this iteration's 2 tiles landed
      } else {
        __builtin_amdgcn_s_wait_tensorcnt(0);
      }
    }
    __syncthreads();                            // publish buf to all waves
    wmma_tile_step(&As[buf][0], &Bs[buf][0], wave_m, wave_n, nlo, hi, acc);
    __syncthreads();                            // done reading buf before overwrite
  }
#else
  // Fallback: manual batched b128 copies (both tiles are plain row-major now)
  for (int k0 = 0; k0 < K; k0 += 32) {
    __syncthreads();
    {
      v4u av[2], bv2[2];
      int row[2], seg[2];
#pragma unroll
      for (int i = 0; i < 2; ++i) {
        int idx = tid + 256 * i;          // 512 segments of 8 halves per tile
        row[i] = idx >> 2;
        seg[i] = (idx & 3) * 8;
        av[i]  = *(const v4u*)(A  + (size_t)(gm0 + row[i]) * K + k0 + seg[i]);
        bv2[i] = *(const v4u*)(BT + (size_t)(gn0 + row[i]) * K + k0 + seg[i]);
      }
#pragma unroll
      for (int i = 0; i < 2; ++i) {
        *(v4u*)(&As[0][row[i] * 32 + seg[i]]) = av[i];
        *(v4u*)(&Bs[0][row[i] * 32 + seg[i]]) = bv2[i];
      }
    }
    if (k0 + 32 < K)
      __builtin_prefetch(BT + (size_t)(gn0 + (tid >> 1)) * K + k0 + 32 + (tid & 1) * 16, 0, 1);
    __syncthreads();
    wmma_tile_step(&As[0][0], &Bs[0][0], wave_m, wave_n, nlo, hi, acc);
  }
#endif

  // --- epilogue: C layout row = r + 8*hi, col = nlo (ISA 7.12.2)
#pragma unroll
  for (int nt = 0; nt < 2; ++nt) {
    int col = gn0 + wave_n + nt * 16 + nlo;
    float bval = bias ? bias[col] : 0.f;
#pragma unroll
    for (int mt = 0; mt < 4; ++mt) {
#pragma unroll
      for (int r = 0; r < 8; ++r) {
        int row = gm0 + wave_m + mt * 16 + r + 8 * hi;
        float val = acc[mt][nt][r] + bval;
        size_t o = (size_t)row * N + col;
        if (epi == 0) {
          out16[o] = (_Float16)val;
        } else if (epi == 1) {
          float g = 0.5f * val * (1.f + erff(val * 0.70710678f));  // exact GELU
          out16[o] = (_Float16)g;
        } else {
          out32[o] = val + resid[o];
        }
      }
    }
  }
}

// ---------------------------------------------------------------------------
// Sliding-window (banded) attention, flash-style online softmax.
// One WG = 128 threads = 4 waves; each wave owns 16 queries.
// Grid: B*H*NB*(W_/64) workgroups. Streams 24 chunks of 32 keys.
// q,k,v layouts: [B,S,H,HD] f16 (exactly what the QKV GEMM produced).
// out: [B,S,D] f16 (head-interleaved, ready for the O-projection GEMM).
// ---------------------------------------------------------------------------
__global__ __launch_bounds__(128) void sw_attn_kernel(
    const _Float16* __restrict__ q, const _Float16* __restrict__ k,
    const _Float16* __restrict__ v, const float* __restrict__ amask,
    const float* __restrict__ hmask, const unsigned char* __restrict__ qmasked,
    _Float16* __restrict__ out) {
  __shared__ _Float16 Pl[4 * 16 * 32];  // per-wave 16x32 probability tile

  const int tid  = threadIdx.x;
  const int lane = tid & 31;
  const int wv   = tid >> 5;
  const int nlo  = lane & 15;
  const int hi   = lane >> 4;

  int t = blockIdx.x;
  const int qs   = t & 3;          t >>= 2;
  const int nb   = t & (NB_ - 1);  t >>= 4;
  const int hidx = t & (H_ - 1);   t >>= 4;
  const int b    = t;

  const int   qloc0 = qs * 64 + wv * 16;          // query offset within block [0,256)
  const int   keybase = nb * W_ - W_;             // absolute index of chunk-local key 0
  const float hm = hmask[hidx];
  const int   wvbase = wv * 16 * 32;

  // Q fragments: A rows = queries (lane row m = nlo), 2 K-steps over hd=64
  const _Float16* qp = q + (((size_t)b * S_ + (nb * W_ + qloc0 + nlo)) * H_ + hidx) * HD_;
  v16h aq0, aq1;
#pragma unroll
  for (int h = 0; h < 16; ++h) {
    int kk = kmapA(h, hi);
    aq0[h] = qp[kk];
    aq1[h] = qp[32 + kk];
  }

  const v8f zf = {0.f, 0.f, 0.f, 0.f, 0.f, 0.f, 0.f, 0.f};
  v8f o[4];
#pragma unroll
  for (int tt = 0; tt < 4; ++tt) o[tt] = zf;
  float mrow[8], lrow[8];
#pragma unroll
  for (int r = 0; r < 8; ++r) { mrow[r] = -1e30f; lrow[r] = 0.f; }

  for (int kc = 0; kc < 3 * W_; kc += 32) {
    // ---- scores: 2 column tiles x 2 hd-steps of WMMA (contraction over hd)
    v8f s[2] = {zf, zf};
#pragma unroll
    for (int tn = 0; tn < 2; ++tn) {
      const int key = keybase + kc + 16 * tn + nlo;  // B-operand col n = nlo
      const bool kin = (key >= 0) && (key < S_);
      const _Float16* kp = kin ? (k + (((size_t)b * S_ + key) * H_ + hidx) * HD_) : nullptr;
#pragma unroll
      for (int ks = 0; ks < 2; ++ks) {
        v16h bk;
#pragma unroll
        for (int h = 0; h < 16; ++h) {
          int hd = ks * 32 + 16 * hi + h;            // contraction index
          bk[h] = kin ? kp[hd] : (_Float16)0.f;
        }
        s[tn] = __builtin_amdgcn_wmma_f32_16x16x32_f16(
            false, ks ? aq1 : aq0, false, bk, (short)0, s[tn], false, false);
      }
    }

    // ---- mask + online softmax (rows live across 16-lane halves; r = row%8)
#pragma unroll
    for (int r = 0; r < 8; ++r) {
      const int qi = qloc0 + r + 8 * hi;  // query index within block [0,256)
      float p0, p1;
      {
        int ki = kc + nlo;                 // chunk-local key, tile 0
        int dlt = ki - W_ - qi;
        int ka = nb * W_ + ki - W_;
        float add = (ka >= 0 && ka < S_) ? amask[b * S_ + ka] : NEGF;
        float sv = s[0][r] * 0.125f + add; // 1/sqrt(64)
        p0 = (dlt >= -W_ && dlt <= W_) ? sv : NEGF;
      }
      {
        int ki = kc + 16 + nlo;            // tile 1
        int dlt = ki - W_ - qi;
        int ka = nb * W_ + ki - W_;
        float add = (ka >= 0 && ka < S_) ? amask[b * S_ + ka] : NEGF;
        float sv = s[1][r] * 0.125f + add;
        p1 = (dlt >= -W_ && dlt <= W_) ? sv : NEGF;
      }
      // row max over the 32 keys: reduce within each 16-lane half
      float cm = fmaxf(p0, p1);
#pragma unroll
      for (int offs = 1; offs < 16; offs <<= 1) cm = fmaxf(cm, __shfl_xor(cm, offs));
      float mnew = fmaxf(mrow[r], cm);
      float sc = __expf(mrow[r] - mnew);
      p0 = __expf(p0 - mnew);
      p1 = __expf(p1 - mnew);
      float ps = p0 + p1;
#pragma unroll
      for (int offs = 1; offs < 16; offs <<= 1) ps += __shfl_xor(ps, offs);
      lrow[r] = lrow[r] * sc + ps;
      mrow[r] = mnew;
#pragma unroll
      for (int tt = 0; tt < 4; ++tt) o[tt][r] *= sc;
      // stash P (C layout) into LDS for re-layout into the A-operand format
      Pl[wvbase + (r + 8 * hi) * 32 + nlo]      = (_Float16)p0;
      Pl[wvbase + (r + 8 * hi) * 32 + 16 + nlo] = (_Float16)p1;
    }
    asm volatile("s_wait_dscnt 0" ::: "memory");  // cross-lane LDS round trip

    // ---- P as A-operand (rows = queries, contraction = 32 keys)
    v16h pa;
#pragma unroll
    for (int h = 0; h < 16; ++h) pa[h] = Pl[wvbase + nlo * 32 + kmapA(h, hi)];

    // ---- O += P * V : B-operand contraction = keys, col n = hd
#pragma unroll
    for (int tt = 0; tt < 4; ++tt) {
      v16h bvv;
#pragma unroll
      for (int h = 0; h < 16; ++h) {
        int key = keybase + kc + 16 * hi + h;
        bvv[h] = (key >= 0 && key < S_)
                     ? v[(((size_t)b * S_ + key) * H_ + hidx) * HD_ + 16 * tt + nlo]
                     : (_Float16)0.f;
      }
      o[tt] = __builtin_amdgcn_wmma_f32_16x16x32_f16(
          false, pa, false, bvv, (short)0, o[tt], false, false);
    }
  }

  // ---- finalize: 1/l, head mask, query mask; write [B,S,H*HD] f16
#pragma unroll
  for (int r = 0; r < 8; ++r) {
    float inv = (lrow[r] > 0.f) ? (hm / lrow[r]) : 0.f;
    int qa = nb * W_ + qloc0 + r + 8 * hi;
    if (qmasked[b * S_ + qa]) inv = 0.f;
    size_t obase = ((size_t)b * S_ + qa) * D_ + hidx * HD_;
#pragma unroll
    for (int tt = 0; tt < 4; ++tt)
      out[obase + 16 * tt + nlo] = (_Float16)(o[tt][r] * inv);
  }
}

// ---------------------------------------------------------------------------
// Row LayerNorm over D_=1024; optional f16 side-copy for the next GEMM.
// ---------------------------------------------------------------------------
__global__ __launch_bounds__(256) void ln_kernel(const float* __restrict__ in,
                                                 const float* __restrict__ g,
                                                 const float* __restrict__ be,
                                                 float* __restrict__ o32,
                                                 _Float16* __restrict__ o16) {
  __shared__ float red[256];
  const int row = blockIdx.x;
  const int tid = threadIdx.x;
  const float* x = in + (size_t)row * D_;

  float s = 0.f;
  for (int c = tid; c < D_; c += 256) s += x[c];
  red[tid] = s;
  __syncthreads();
  for (int st = 128; st > 0; st >>= 1) {
    if (tid < st) red[tid] += red[tid + st];
    __syncthreads();
  }
  float mean = red[0] * (1.f / D_);
  __syncthreads();

  float s2 = 0.f;
  for (int c = tid; c < D_; c += 256) {
    float d = x[c] - mean;
    s2 += d * d;
  }
  red[tid] = s2;
  __syncthreads();
  for (int st = 128; st > 0; st >>= 1) {
    if (tid < st) red[tid] += red[tid + st];
    __syncthreads();
  }
  float rstd = rsqrtf(red[0] * (1.f / D_) + 1e-5f);

  for (int c = tid; c < D_; c += 256) {
    float yv = (x[c] - mean) * rstd * g[c] + be[c];
    o32[(size_t)row * D_ + c] = yv;
    if (o16) o16[(size_t)row * D_ + c] = (_Float16)yv;
  }
}

// ---------------------------------------------------------------------------
// Host-side orchestration
// ---------------------------------------------------------------------------
extern "C" void kernel_launch(void* const* d_in, const int* in_sizes, int n_in,
                              void* d_out, int out_size, void* d_ws, size_t ws_size,
                              hipStream_t stream) {
  (void)in_sizes; (void)n_in; (void)out_size; (void)ws_size;

  const float* hs    = (const float*)d_in[0];
  const float* amask = (const float*)d_in[1];
  const float* hmask = (const float*)d_in[2];
  const unsigned char* qmasked = (const unsigned char*)d_in[3];
  // d_in[4] is_index_global_attn, d_in[5] is_global_attn: zero in this config
  const float* Wq = (const float*)d_in[6];  const float* bq = (const float*)d_in[7];
  const float* Wk = (const float*)d_in[8];  const float* bk = (const float*)d_in[9];
  const float* Wv = (const float*)d_in[10]; const float* bv = (const float*)d_in[11];
  const float* Wo = (const float*)d_in[12]; const float* bo = (const float*)d_in[13];
  const float* g1 = (const float*)d_in[14]; const float* be1 = (const float*)d_in[15];
  const float* W1 = (const float*)d_in[16]; const float* b1 = (const float*)d_in[17];
  const float* W2 = (const float*)d_in[18]; const float* b2 = (const float*)d_in[19];
  const float* g2 = (const float*)d_in[20]; const float* be2 = (const float*)d_in[21];

  char* ws = (char*)d_ws;
  size_t off = 0;
  auto wsalloc = [&](size_t bytes) -> char* {
    char* p = ws + off;
    off = (off + bytes + 255) & ~(size_t)255;
    return p;
  };
  const size_t BSD = (size_t)B_ * S_ * D_;      // 8,388,608
  const size_t BSF = (size_t)B_ * S_ * FFN_;    // 33,554,432
  const size_t DD  = (size_t)D_ * D_;
  const size_t DF  = (size_t)D_ * FFN_;

  _Float16* hs16   = (_Float16*)wsalloc(BSD * 2);
  _Float16* wqT    = (_Float16*)wsalloc(DD * 2);   // [N,K] transposed f16 weights
  _Float16* wkT    = (_Float16*)wsalloc(DD * 2);
  _Float16* wvT    = (_Float16*)wsalloc(DD * 2);
  _Float16* woT    = (_Float16*)wsalloc(DD * 2);
  _Float16* w1T    = (_Float16*)wsalloc(DF * 2);
  _Float16* w2T    = (_Float16*)wsalloc(DF * 2);
  _Float16* q16    = (_Float16*)wsalloc(BSD * 2);
  _Float16* k16    = (_Float16*)wsalloc(BSD * 2);
  _Float16* v16    = (_Float16*)wsalloc(BSD * 2);
  _Float16* attn16 = (_Float16*)wsalloc(BSD * 2);
  float*    xpre   = (float*)wsalloc(BSD * 4);
  float*    x32    = (float*)wsalloc(BSD * 4);
  _Float16* x16    = (_Float16*)wsalloc(BSD * 2);
  _Float16* h16    = (_Float16*)wsalloc(BSF * 2);
  float*    y2     = (float*)wsalloc(BSD * 4);

  const int M = B_ * S_;  // 8192

  // 1) fp16 conversions: hs layout-preserving; weights transpose-converted
  cvt_f16_kernel<<<(int)((BSD + 1023) / 1024), 256, 0, stream>>>(hs, hs16, (int)BSD);
  dim3 tDD(D_ / 32, D_ / 32);
  cvt_t_f16_kernel<<<tDD, 256, 0, stream>>>(Wq, wqT, D_, D_);
  cvt_t_f16_kernel<<<tDD, 256, 0, stream>>>(Wk, wkT, D_, D_);
  cvt_t_f16_kernel<<<tDD, 256, 0, stream>>>(Wv, wvT, D_, D_);
  cvt_t_f16_kernel<<<tDD, 256, 0, stream>>>(Wo, woT, D_, D_);
  dim3 tW1(D_ / 32, FFN_ / 32);
  cvt_t_f16_kernel<<<tW1, 256, 0, stream>>>(W1, w1T, D_, FFN_);
  dim3 tW2(FFN_ / 32, D_ / 32);
  cvt_t_f16_kernel<<<tW2, 256, 0, stream>>>(W2, w2T, FFN_, D_);

  // 2) QKV projections
  dim3 gDD(M / 128, D_ / 128);
  gemm_f16_kernel<<<gDD, 256, 0, stream>>>(hs16, wqT, bq, nullptr, q16, nullptr, M, D_, D_, 0);
  gemm_f16_kernel<<<gDD, 256, 0, stream>>>(hs16, wkT, bk, nullptr, k16, nullptr, M, D_, D_, 0);
  gemm_f16_kernel<<<gDD, 256, 0, stream>>>(hs16, wvT, bv, nullptr, v16, nullptr, M, D_, D_, 0);

  // 3) banded attention
  sw_attn_kernel<<<B_ * H_ * NB_ * (W_ / 64), 128, 0, stream>>>(
      q16, k16, v16, amask, hmask, qmasked, attn16);

  // 4) output projection + residual -> xpre (f32)
  gemm_f16_kernel<<<gDD, 256, 0, stream>>>(attn16, woT, bo, hs, nullptr, xpre, M, D_, D_, 2);

  // 5) LN1 -> x32 (f32) + x16 (f16)
  ln_kernel<<<M, 256, 0, stream>>>(xpre, g1, be1, x32, x16);

  // 6) FFN1 + exact GELU -> h16
  dim3 gF1(M / 128, FFN_ / 128);
  gemm_f16_kernel<<<gF1, 256, 0, stream>>>(x16, w1T, b1, nullptr, h16, nullptr, M, FFN_, D_, 1);

  // 7) FFN2 + residual(x32) -> y2 (f32)
  gemm_f16_kernel<<<gDD, 256, 0, stream>>>(h16, w2T, b2, x32, nullptr, y2, M, D_, FFN_, 2);

  // 8) LN2 -> final fp32 output
  ln_kernel<<<M, 256, 0, stream>>>(y2, g2, be2, (float*)d_out, nullptr);
}
